// GatedScaledAttention_25469156065361
// MI455X (gfx1250) — matrix operand order repaired and checked by reading, not compiled, not run on hardware
//
#include <hip/hip_runtime.h>
#include <hip/hip_bf16.h>

typedef __attribute__((ext_vector_type(16))) _Float16 v16h;
typedef __attribute__((ext_vector_type(8)))  _Float16 v8h;
typedef __attribute__((ext_vector_type(8)))  float    v8f;

#define D_MODEL 1024
#define NH      16
#define HD      64
#define TSEQ    1024
#define BATCH   4
#define MROWS   (BATCH * TSEQ)       // 4096
#define NEG_BIG (-1.0e30f)

// ---------------------------------------------------------------- helpers
static __device__ __forceinline__ v8f vzero8() {
  v8f z;
#pragma unroll
  for (int i = 0; i < 8; ++i) z[i] = 0.0f;
  return z;
}

// 16 halves from two 16-byte chunks (both 16B aligned)
static __device__ __forceinline__ v16h load_frag(const _Float16* p0, const _Float16* p1) {
  v8h a = *(const v8h*)p0;
  v8h b = *(const v8h*)p1;
  v16h r;
#pragma unroll
  for (int i = 0; i < 8; ++i) { r[i] = a[i]; r[i + 8] = b[i]; }
  return r;
}

// exchange with lane^16 (wave32) via ds_bpermute
static __device__ __forceinline__ float xor16f(float v) {
  int idx = (((threadIdx.x & 31) ^ 16) << 2);
  return __int_as_float(__builtin_amdgcn_ds_bpermute(idx, __float_as_int(v)));
}

static __device__ __forceinline__ v8f wmma_f16(v16h a, v16h b, v8f c) {
  return __builtin_amdgcn_wmma_f32_16x16x32_f16(false, a, false, b, (short)0, c, false, false);
}

// gfx1250 async global->LDS copy, 16B per lane, tracked by ASYNCcnt.
// VDST = VGPR holding LDS byte address; VADDR = 64-bit global address.
static __device__ __forceinline__ void async_ldg_b128(void* lds, const void* g) {
  unsigned lds_off = (unsigned)(unsigned long long)(uintptr_t)lds;
  unsigned long long ga = (unsigned long long)(uintptr_t)g;
  asm volatile("global_load_async_to_lds_b128 %0, %1, off"
               :: "v"(lds_off), "v"(ga)
               : "memory");
}

static __device__ __forceinline__ void wait_async0() {
#if __has_builtin(__builtin_amdgcn_s_wait_asynccnt)
  __builtin_amdgcn_s_wait_asynccnt(0);
#else
  asm volatile("s_wait_asynccnt 0x0" ::: "memory");
#endif
}

// ---------------------------------------------------------------- small kernels
__global__ void cvt_f16_kernel(const float* __restrict__ x, _Float16* __restrict__ y, int n) {
  int i = blockIdx.x * 256 + threadIdx.x;
  if (i < n) y[i] = (_Float16)x[i];
}

// Pack W[N=out][K=in] (row-major f32) into B-fragment tiles:
// tile = kt*64 + nt (kt: K/32, nt: N/16); within tile: lane(0..31) x 16 halves.
// value(lane,h) = W[nt*16 + lane%16][kt*32 + (lane<16 ? h : 16+h)]
__global__ void pack_w_kernel(const float* __restrict__ W, _Float16* __restrict__ Wp) {
  int idx  = blockIdx.x * 256 + threadIdx.x;       // < 1024*1024
  int tile = idx >> 9;
  int rem  = idx & 511;
  int ln   = rem >> 4;
  int h    = rem & 15;
  int kt = tile >> 6, nt = tile & 63;
  int n = nt * 16 + (ln & 15);
  int k = kt * 32 + ((ln < 16) ? h : 16 + h);
  Wp[idx] = (_Float16)W[n * D_MODEL + k];
}

// allow/has_valid from key_padding_mask (int 0/1 per element)
__global__ void maskaux_kernel(const int* __restrict__ mask,
                               float* __restrict__ allowf, float* __restrict__ hasv) {
  __shared__ int allpad;
  int b = blockIdx.x;
  if (threadIdx.x == 0) allpad = 1;
  __syncthreads();
  for (int i = threadIdx.x; i < TSEQ; i += 256)
    if (mask[b * TSEQ + i] == 0) atomicAnd(&allpad, 0);
  __syncthreads();
  int ap = allpad;
  for (int i = threadIdx.x; i < TSEQ; i += 256)
    allowf[b * TSEQ + i] = ap ? 1.0f : (mask[b * TSEQ + i] ? 0.0f : 1.0f);
  if (threadIdx.x == 0) hasv[b] = ap ? 0.0f : 1.0f;
}

// RoPE: Y f32 [B*T, 1024] -> out f16 [B,H,T,64]; one thread per (row, head, pair)
__global__ void rope_kernel(const float* __restrict__ Y, _Float16* __restrict__ outh) {
  int idx = blockIdx.x * 256 + threadIdx.x;        // < 4096*512
  int row = idx >> 9;
  int p   = idx & 511;
  int hd  = p >> 5;
  int i   = p & 31;
  int b = row >> 10, t = row & 1023;
  float freq = __expf(-0.14391156f * (float)(2 * i));   // -ln(10000)/64
  float ang  = (float)t * freq;
  float sn, cs;
  sincosf(ang, &sn, &cs);
  const float* src = Y + (size_t)row * D_MODEL + hd * HD + 2 * i;
  float e = src[0], o = src[1];
  _Float16* dst = outh + (((size_t)(b * NH + hd)) * TSEQ + t) * HD + 2 * i;
  dst[0] = (_Float16)(e * cs - o * sn);
  dst[1] = (_Float16)(e * sn + o * cs);
}

// V transpose: Y f32 [B*T,1024] -> vt f16 [B,H,64,T]
__global__ void vtrans_kernel(const float* __restrict__ Y, _Float16* __restrict__ vt) {
  int idx = blockIdx.x * 256 + threadIdx.x;        // < 4096*1024
  int row = idx >> 10, col = idx & 1023;
  int hd = col >> 6, d = col & 63;
  int b = row >> 10, t = row & 1023;
  vt[(((size_t)(b * NH + hd)) * HD + d) * TSEQ + t] = (_Float16)Y[idx];
}

// gated = attn * sigmoid(gate_proj)
__global__ void gate_kernel(const _Float16* __restrict__ attn, const float* __restrict__ Yg,
                            _Float16* __restrict__ gated, int n) {
  int i = blockIdx.x * 256 + threadIdx.x;
  if (i < n) {
    float g = 1.0f / (1.0f + __expf(-Yg[i]));
    gated[i] = (_Float16)((float)attn[i] * g);
  }
}

// ---------------------------------------------------------------- GEMM (Y = X @ W^T + b)
// X: f16 [M,1024] row-major; Wp: packed B-fragment tiles; Y: f32 [M,1024]
// Block = 8 waves (2m x 4n), block tile 64m x 256n, per-wave 32x64.
// A (64x32) and B (32x256, fragment-packed => contiguous 16KB) tiles are staged
// into LDS with async global->LDS b128 copies, double buffered on ASYNCcnt.
__global__ __launch_bounds__(256) void gemm16_kernel(const _Float16* __restrict__ X,
                                                     const _Float16* __restrict__ Wp,
                                                     const float* __restrict__ bias,
                                                     float* __restrict__ Y, int M) {
  __shared__ _Float16 sA[2][64 * 32];      // 2 x 4KB
  __shared__ _Float16 sB[2][16 * 512];     // 2 x 16KB

  int tid  = threadIdx.x;
  int wave = tid >> 5;
  int lane = tid & 31;
  int ln = lane & 15, hi = lane >> 4;

  int blockM = (blockIdx.x >> 2) * 64;
  int blockN = (blockIdx.x & 3) * 256;
  if (blockM >= M) return;
  int mi2 = wave >> 2, ni2 = wave & 3;     // wave position in 2m x 4n grid
  int ntb = blockN >> 4;                   // first packed-tile column of this block

  int arow = tid >> 2, aseg = tid & 3;     // A staging assignment (16B per thread)

  auto stage = [&](int kt, int buf) {
    const _Float16* bsrc = Wp + ((size_t)(kt * 64) + ntb) * 512;
#pragma unroll
    for (int c = 0; c < 4; ++c) {
      int e = (tid + c * 256) * 8;         // halves
      async_ldg_b128(&sB[buf][e], bsrc + e);
    }
    const _Float16* asrc = X + (size_t)(blockM + arow) * D_MODEL + kt * 32 + aseg * 8;
    async_ldg_b128(&sA[buf][arow * 32 + aseg * 8], asrc);
  };

  v8f c[2][4];
#pragma unroll
  for (int mi = 0; mi < 2; ++mi)
#pragma unroll
    for (int ni = 0; ni < 4; ++ni) c[mi][ni] = vzero8();

  stage(0, 0);
  wait_async0();
  __syncthreads();

  for (int kt = 0; kt < 32; ++kt) {
    int cur = kt & 1;
    if (kt + 1 < 32) stage(kt + 1, cur ^ 1);

    v16h a[2];
#pragma unroll
    for (int mi = 0; mi < 2; ++mi) {
      const _Float16* p = &sA[cur][(mi2 * 32 + mi * 16 + ln) * 32 + 16 * hi];
      a[mi] = load_frag(p, p + 8);
    }
    v16h bf[4];
#pragma unroll
    for (int ni = 0; ni < 4; ++ni) {
      const _Float16* q = &sB[cur][(ni2 * 4 + ni) * 512 + lane * 16];
      bf[ni] = load_frag(q, q + 8);
    }
#pragma unroll
    for (int mi = 0; mi < 2; ++mi)
#pragma unroll
      for (int ni = 0; ni < 4; ++ni)
        c[mi][ni] = wmma_f16(a[mi], bf[ni], c[mi][ni]);

    wait_async0();          // next buffer resident (this wave's share)
    __syncthreads();        // all waves' shares resident; reads of cur complete
  }

#pragma unroll
  for (int ni = 0; ni < 4; ++ni) {
    int n = blockN + ni2 * 64 + ni * 16 + ln;
    float bv = bias[n];
#pragma unroll
    for (int mi = 0; mi < 2; ++mi)
#pragma unroll
      for (int r = 0; r < 8; ++r) {
        int m = blockM + mi2 * 32 + mi * 16 + r + 8 * hi;
        Y[(size_t)m * D_MODEL + n] = c[mi][ni][r] + bv;
      }
  }
}

// ---------------------------------------------------------------- flash attention (transposed)
// qh,kh: f16 [B,H,T,64]; vt: f16 [B,H,64,T]; out: f16 [B*T, 1024]
// Wave owns 16 query rows; computes S^T = K·Q^T so each lane owns one q column.
__global__ __launch_bounds__(256) void attn_kernel(const _Float16* __restrict__ qh,
                                                   const _Float16* __restrict__ kh,
                                                   const _Float16* __restrict__ vt,
                                                   const float* __restrict__ allowf,
                                                   const float* __restrict__ hasv,
                                                   _Float16* __restrict__ out) {
  int wave = (blockIdx.x * 256 + threadIdx.x) >> 5;   // < 4096
  int lane = threadIdx.x & 31;
  int ln = lane & 15, hi = lane >> 4;
  int qt = wave & 63;
  int hd = (wave >> 6) & 15;
  int b  = wave >> 10;
  int q0 = qt * 16;

  // Q as B-fragments (lane = q column, halves = d), persistent across key loop
  const _Float16* qbase = qh + (((size_t)(b * NH + hd)) * TSEQ + q0 + ln) * HD;
  v16h qb[2];
#pragma unroll
  for (int d0i = 0; d0i < 2; ++d0i) {
    const _Float16* p = qbase + d0i * 32 + hi * 16;
    qb[d0i] = load_frag(p, p + 8);
  }

  const _Float16* kbase = kh + ((size_t)(b * NH + hd)) * TSEQ * HD;
  const _Float16* vbase = vt + ((size_t)(b * NH + hd)) * HD * TSEQ;
  const float*    al    = allowf + b * TSEQ;

  v8f Oacc[4];
#pragma unroll
  for (int dt = 0; dt < 4; ++dt) Oacc[dt] = vzero8();
  float mrun = -3.0e38f, lrun = 0.0f;

  for (int tk0 = 0; tk0 < TSEQ; tk0 += 32) {
    // S^T tiles: rows tk (2 tiles of 16), cols q
    v8f s[2];
#pragma unroll
    for (int t = 0; t < 2; ++t) {
      v8f cc = vzero8();
      const _Float16* krow = kbase + (size_t)(tk0 + 16 * t + ln) * HD;
#pragma unroll
      for (int d0i = 0; d0i < 2; ++d0i) {
        const _Float16* p = krow + d0i * 32 + 16 * hi;
        v16h ka = load_frag(p, p + 8);
        cc = wmma_f16(ka, qb[d0i], cc);
      }
      s[t] = cc;
    }

    // scale + key-padding mask (where-style), block max
    float p0v[8], p1v[8];
    float mx = -3.0e38f;
#pragma unroll
    for (int r = 0; r < 8; ++r) {
      int tkA = tk0 + r + 8 * hi;
      int tkB = tk0 + 16 + r + 8 * hi;
      float v0 = (al[tkA] > 0.5f) ? s[0][r] * 0.125f : NEG_BIG;
      float v1 = (al[tkB] > 0.5f) ? s[1][r] * 0.125f : NEG_BIG;
      p0v[r] = v0; p1v[r] = v1;
      mx = fmaxf(mx, fmaxf(v0, v1));
    }
    mx = fmaxf(mx, xor16f(mx));            // combine lane pair sharing this q column
    float mnew  = fmaxf(mrun, mx);
    float alpha = __expf(mrun - mnew);
    float rs = 0.0f;
#pragma unroll
    for (int r = 0; r < 8; ++r) {
      p0v[r] = __expf(p0v[r] - mnew);
      p1v[r] = __expf(p1v[r] - mnew);
      rs += p0v[r] + p1v[r];
    }
    rs += xor16f(rs);
    lrun = lrun * alpha + rs;
    mrun = mnew;
#pragma unroll
    for (int dt = 0; dt < 4; ++dt)
#pragma unroll
      for (int r = 0; r < 8; ++r) Oacc[dt][r] *= alpha;

    // Build P^T B-fragment (lane = q, halves = tk) with one xor-16 half swap
    v16h pb;
#pragma unroll
    for (int r = 0; r < 8; ++r) {
      float o0 = xor16f(p0v[r]);
      float o1 = xor16f(p1v[r]);
      pb[r]     = (_Float16)(hi ? o1 : p0v[r]);
      pb[r + 8] = (_Float16)(hi ? p1v[r] : o0);
    }

    // O^T += V^T · P^T  (A: lane = d, halves = tk; contiguous from vt)
#pragma unroll
    for (int dt = 0; dt < 4; ++dt) {
      const _Float16* vrow = vbase + (size_t)(dt * 16 + ln) * TSEQ + tk0 + 16 * hi;
      v16h va = load_frag(vrow, vrow + 8);
      Oacc[dt] = wmma_f16(va, pb, Oacc[dt]);
    }
  }

  float sc = hasv[b] / lrun;
  _Float16* orow = out + ((size_t)b * TSEQ + q0 + ln) * D_MODEL + hd * HD;
#pragma unroll
  for (int dt = 0; dt < 4; ++dt)
#pragma unroll
    for (int r = 0; r < 8; ++r) {
      int d = dt * 16 + r + 8 * hi;
      orow[d] = (_Float16)(Oacc[dt][r] * sc);
    }
}

// ---------------------------------------------------------------- launch
extern "C" void kernel_launch(void* const* d_in, const int* in_sizes, int n_in,
                              void* d_out, int out_size, void* d_ws, size_t ws_size,
                              hipStream_t stream) {
  const float* query = (const float*)d_in[0];
  const float* key   = (const float*)d_in[1];
  const float* value = (const float*)d_in[2];
  const int*   mask  = (const int*)d_in[3];
  const float* Wq = (const float*)d_in[4];  const float* bq = (const float*)d_in[5];
  const float* Wk = (const float*)d_in[6];  const float* bk = (const float*)d_in[7];
  const float* Wv = (const float*)d_in[8];  const float* bv = (const float*)d_in[9];
  const float* Wo = (const float*)d_in[10]; const float* bo = (const float*)d_in[11];
  const float* Wg = (const float*)d_in[12]; const float* bg = (const float*)d_in[13];
  float* outp = (float*)d_out;

  const size_t ND = (size_t)MROWS * D_MODEL;          // 4M elems
  char* wsp = (char*)d_ws;
  size_t off = 0;
  auto carve = [&](size_t bytes) -> void* {
    void* p = wsp + off;
    off += (bytes + 255) & ~(size_t)255;
    return p;
  };
  _Float16* x16q = (_Float16*)carve(ND * 2);
  _Float16* x16k = (_Float16*)carve(ND * 2);
  _Float16* x16v = (_Float16*)carve(ND * 2);
  _Float16* wqp  = (_Float16*)carve((size_t)D_MODEL * D_MODEL * 2);
  _Float16* wkp  = (_Float16*)carve((size_t)D_MODEL * D_MODEL * 2);
  _Float16* wvp  = (_Float16*)carve((size_t)D_MODEL * D_MODEL * 2);
  _Float16* wgp  = (_Float16*)carve((size_t)D_MODEL * D_MODEL * 2);
  _Float16* wop  = (_Float16*)carve((size_t)D_MODEL * D_MODEL * 2);
  float*    Y0   = (float*)carve(ND * 4);
  float*    Yg   = (float*)carve(ND * 4);
  _Float16* qh16 = (_Float16*)carve(ND * 2);
  _Float16* kh16 = (_Float16*)carve(ND * 2);
  _Float16* vt16 = (_Float16*)carve(ND * 2);
  _Float16* at16 = (_Float16*)carve(ND * 2);
  _Float16* gd16 = (_Float16*)carve(ND * 2);
  float*    alw  = (float*)carve((size_t)BATCH * TSEQ * 4);
  float*    hv   = (float*)carve((size_t)BATCH * 4);

  const int CVT_BLK  = (int)(ND / 256);               // 16384
  const int PACK_BLK = D_MODEL * D_MODEL / 256;       // 4096
  const int GEMM_BLK = (MROWS / 64) * (D_MODEL / 256);      // 256
  const int ROPE_BLK = MROWS * 512 / 256;             // 8192
  const int ATTN_BLK = (BATCH * NH * (TSEQ / 16)) / 8;      // 512

  // weights -> packed f16 fragments
  pack_w_kernel<<<PACK_BLK, 256, 0, stream>>>(Wq, wqp);
  pack_w_kernel<<<PACK_BLK, 256, 0, stream>>>(Wk, wkp);
  pack_w_kernel<<<PACK_BLK, 256, 0, stream>>>(Wv, wvp);
  pack_w_kernel<<<PACK_BLK, 256, 0, stream>>>(Wg, wgp);
  pack_w_kernel<<<PACK_BLK, 256, 0, stream>>>(Wo, wop);
  // activations -> f16
  cvt_f16_kernel<<<CVT_BLK, 256, 0, stream>>>(query, x16q, (int)ND);
  cvt_f16_kernel<<<CVT_BLK, 256, 0, stream>>>(key,   x16k, (int)ND);
  cvt_f16_kernel<<<CVT_BLK, 256, 0, stream>>>(value, x16v, (int)ND);
  // padding-mask aux
  maskaux_kernel<<<BATCH, 256, 0, stream>>>(mask, alw, hv);

  // q proj + RoPE
  gemm16_kernel<<<GEMM_BLK, 256, 0, stream>>>(x16q, wqp, bq, Y0, MROWS);
  rope_kernel<<<ROPE_BLK, 256, 0, stream>>>(Y0, qh16);
  // k proj + RoPE
  gemm16_kernel<<<GEMM_BLK, 256, 0, stream>>>(x16k, wkp, bk, Y0, MROWS);
  rope_kernel<<<ROPE_BLK, 256, 0, stream>>>(Y0, kh16);
  // v proj + transpose
  gemm16_kernel<<<GEMM_BLK, 256, 0, stream>>>(x16v, wvp, bv, Y0, MROWS);
  vtrans_kernel<<<CVT_BLK, 256, 0, stream>>>(Y0, vt16);
  // gate proj
  gemm16_kernel<<<GEMM_BLK, 256, 0, stream>>>(x16q, wgp, bg, Yg, MROWS);

  // flash attention (register-resident, WMMA)
  attn_kernel<<<ATTN_BLK, 256, 0, stream>>>(qh16, kh16, vt16, alw, hv, at16);

  // gating + output projection (f32 result)
  gate_kernel<<<CVT_BLK, 256, 0, stream>>>(at16, Yg, gd16, (int)ND);
  gemm16_kernel<<<GEMM_BLK, 256, 0, stream>>>(gd16, wop, bo, outp, MROWS);
}
